// Attn_51694226375472
// MI455X (gfx1250) — compile-verified, matrix operand order checked by reading, not run on hardware
//
#include <hip/hip_runtime.h>

// ---------------------------------------------------------------------------
// Shapes
// ---------------------------------------------------------------------------
#define H   300
#define L   2048
#define B   128
#define H2  600          // 2*H
#define NT  19           // ceil(304/16) N tiles (N padded 300->304)
#define KC  10           // K chunks of 32 (K padded 300->320)
#define BFRAG_DW 4864    // per-K-chunk W2 fragment dwords: NT*32 lanes*8 dw

typedef __attribute__((ext_vector_type(16))) __bf16   v16bf;
typedef __attribute__((ext_vector_type(8)))  float    v8f;
typedef __attribute__((ext_vector_type(4)))  unsigned uint4n;
typedef __attribute__((ext_vector_type(4)))  unsigned u32x4;
typedef __attribute__((ext_vector_type(4)))  int      i32x4;
typedef __attribute__((ext_vector_type(8)))  int      i32x8;

union FragBF { uint4n q[2]; v16bf v; };

#if defined(__has_builtin)
#if __has_builtin(__builtin_amdgcn_tensor_load_to_lds) && \
    __has_builtin(__builtin_amdgcn_s_wait_tensorcnt)
#define HAVE_TDM 1
#endif
#endif
#ifndef HAVE_TDM
#define HAVE_TDM 0
#endif

// Split f32 into bf16 hi + bf16 lo (truncation; x - hi is exact in f32).
__device__ inline void split_bf16(float x, unsigned& h, unsigned& l) {
    unsigned bx = __float_as_uint(x);
    unsigned hb = bx & 0xFFFF0000u;
    float r = x - __uint_as_float(hb);
    h = hb >> 16;
    l = (__float_as_uint(r) & 0xFFFF0000u) >> 16;
}

// tanh via v_exp (TRANS op), clamped so exp never overflows.
__device__ inline float fast_tanh(float x) {
    x = fminf(fmaxf(x, -44.0f), 44.0f);
    float e = __expf(2.0f * x);
    return (e - 1.0f) / (e + 1.0f);
}

#if HAVE_TDM
// ---------------------------------------------------------------------------
// TDM: 1-D DMA copy of `ndw` dwords global -> LDS (ISA 08_async_tensor §8).
// D# group0: [1:0]=count=1, [63:32]=lds byte addr, [120:64]=global addr,
//            [127:126]=type=2.  group1: data_size=4B (code 2), tile_dim0 =
//            tensor_dim0 = ndw, tile_dim1 = tensor_dim1 = 1, stride0 = ndw.
// Wave-scalar op: EXEC ignored, tracked by TENSORcnt.
// ---------------------------------------------------------------------------
__device__ inline void tdm_copy_to_lds(const void* gsrc, unsigned lds_byte_off,
                                       unsigned ndw) {
    unsigned long long ga = (unsigned long long)gsrc;
    u32x4 g0;
    g0[0] = 1u;                                   // count=1, user mode
    g0[1] = lds_byte_off;                         // lds_addr
    g0[2] = (unsigned)(ga & 0xFFFFFFFFu);         // global_addr[31:0]
    g0[3] = (unsigned)((ga >> 32) & 0x01FFFFFFu)  // global_addr[56:32]
          | (2u << 30);                           // type=2 ("image")
    i32x8 g1;
    g1[0] = (int)(2u << 16);                      // wg_mask=0, data_size=4B
    g1[1] = (int)((ndw & 0xFFFFu) << 16);         // tensor_dim0[15:0]
    g1[2] = (int)(((ndw >> 16) & 0xFFFFu) | (1u << 16)); // dim0[31:16], dim1=1
    g1[3] = (int)((ndw & 0xFFFFu) << 16);         // tile_dim0 = ndw
    g1[4] = 1;                                    // tile_dim1=1, tile_dim2=0
    g1[5] = (int)ndw;                             // tensor_dim0_stride[31:0]
    g1[6] = 0;
    g1[7] = 0;
    i32x4 z4 = {0, 0, 0, 0};
#if __clang_major__ >= 23
    i32x8 z8 = {0, 0, 0, 0, 0, 0, 0, 0};
    __builtin_amdgcn_tensor_load_to_lds(g0, g1, z4, z4, z8, 0);
#else
    __builtin_amdgcn_tensor_load_to_lds(g0, g1, z4, z4, 0);
#endif
}
#endif // HAVE_TDM

// ---------------------------------------------------------------------------
// 1) pre[b,n] = attn_b[n] + sum_h lh[b,h] * attn_W[n, h]   (W1 half)
// ---------------------------------------------------------------------------
__global__ __launch_bounds__(320) void pre_kernel(
    const float* __restrict__ lh, const float* __restrict__ aW,
    const float* __restrict__ ab, float* __restrict__ pre)
{
    int b = blockIdx.x, tid = threadIdx.x;
    __shared__ float sh[H];
    if (tid < H) sh[tid] = lh[b * H + tid];
    __syncthreads();
    if (tid < H) {
        float a = ab[tid];
        const float* w = aW + tid * H2;
        #pragma unroll 4
        for (int h = 0; h < H; ++h) a = fmaf(sh[h], w[h], a);
        pre[b * H + tid] = a;
    }
}

// ---------------------------------------------------------------------------
// 2) Pre-swizzle W2^T (K=300 x N=300, padded 320x304) into per-lane WMMA
//    B-operand fragments, split into bf16 hi/lo.
//    Layout: t = ((chunk*NT + nt)*32 + lane)*8 + j ; dword = pk(k0, k0+1).
//    16-bit operand layout (ISA 7.12.2): VGPR j <-> K pair
//    k0 = (j<4 ? 2j : 16+2(j-4)) + (lane>=16 ? 8 : 0).
// ---------------------------------------------------------------------------
__global__ void w2prep_kernel(const float* __restrict__ aW,
                              unsigned* __restrict__ Bh,
                              unsigned* __restrict__ Bl)
{
    int t = blockIdx.x * 256 + threadIdx.x;
    if (t >= KC * BFRAG_DW) return;
    int j    = t & 7;
    int lane = (t >> 3) & 31;
    int rest = t >> 8;
    int nt   = rest % NT;
    int c    = rest / NT;
    int n    = nt * 16 + (lane & 15);
    int half = lane >> 4;
    int k0   = c * 32 + ((j < 4) ? 2 * j : 16 + 2 * (j - 4)) + half * 8;
    // W2^T[k][n] = attn_W[n*2H + H + k]
    float x0 = (n < H && k0     < H) ? aW[n * H2 + H + k0]     : 0.0f;
    float x1 = (n < H && k0 + 1 < H) ? aW[n * H2 + H + k0 + 1] : 0.0f;
    unsigned h0, l0, h1, l1;
    split_bf16(x0, h0, l0);
    split_bf16(x1, h1, l1);
    Bh[t] = h0 | (h1 << 16);
    Bl[t] = l0 | (l1 << 16);
}

// ---------------------------------------------------------------------------
// 3) Fused energy GEMM + tanh + v-dot -> score[b,l].
//    One block per l (128 rows = all b). 8 waves; wave w owns rows
//    [w*16, w*16+16) x all 19 N tiles. Split-bf16: 3 WMMAs per (nt, kchunk).
//    W2 fragments are DMA'd L2->LDS by the Tensor Data Mover (wave 0),
//    overlapping with the VALU f32->bf16hi/lo conversion of the enc tile.
// ---------------------------------------------------------------------------
__global__ __launch_bounds__(256) void energy_score_kernel(
    const float* __restrict__ enc, const float* __restrict__ pre,
    const unsigned* __restrict__ Bh, const unsigned* __restrict__ Bl,
    const float* __restrict__ vvec, float* __restrict__ score)
{
    __shared__ __align__(16) unsigned sA[128 * 32];   // [row][0..15]=hi, [16..31]=lo
    __shared__ __align__(16) unsigned sBh[BFRAG_DW];
    __shared__ __align__(16) unsigned sBl[BFRAG_DW];

    const int tid  = threadIdx.x;
    const int lane = tid & 31;
    const int wave = tid >> 5;
    const int l    = blockIdx.x;
    const long rowBase = (long)l * B;          // flattened [L*B, H] row base
    const int  m    = lane & 15;
    const int  half = lane >> 4;

    v8f acc[NT];
    #pragma unroll
    for (int i = 0; i < NT; ++i)
        #pragma unroll
        for (int j = 0; j < 8; ++j) acc[i][j] = 0.0f;

    for (int c = 0; c < KC; ++c) {
        // --- B staging: TDM DMA (wave 0 issues; generic shared ptr low 32
        //     bits are the LDS byte offset per the aperture mapping) ---
#if HAVE_TDM
        if (wave == 0) {
            tdm_copy_to_lds(Bh + c * BFRAG_DW,
                            (unsigned)(unsigned long long)(void*)sBh, BFRAG_DW);
            tdm_copy_to_lds(Bl + c * BFRAG_DW,
                            (unsigned)(unsigned long long)(void*)sBl, BFRAG_DW);
        }
#else
        {
            const unsigned* bhs = Bh + c * BFRAG_DW;
            const unsigned* bls = Bl + c * BFRAG_DW;
            for (int i = tid; i < BFRAG_DW; i += 256) { sBh[i] = bhs[i]; sBl[i] = bls[i]; }
        }
#endif
        // --- stage A (enc rows, pair-packed hi/lo bf16) — overlaps the DMA ---
        for (int i = tid; i < 2048; i += 256) {
            int row = i >> 4, p = i & 15;
            int k = c * 32 + 2 * p;
            const float* src = enc + (rowBase + row) * H;
            float x0 = (k     < H) ? src[k]     : 0.0f;
            float x1 = (k + 1 < H) ? src[k + 1] : 0.0f;
            unsigned h0, l0, h1, l1;
            split_bf16(x0, h0, l0);
            split_bf16(x1, h1, l1);
            sA[row * 32 + p]      = h0 | (h1 << 16);
            sA[row * 32 + 16 + p] = l0 | (l1 << 16);
        }
#if HAVE_TDM
        if (wave == 0) __builtin_amdgcn_s_wait_tensorcnt(0);
#endif
        __syncthreads();

        // --- A fragments for this wave's 16-row tile ---
        const int abase = (wave * 16 + m) * 32 + half * 4;
        FragBF ahi, alo;
        ahi.q[0] = *(const uint4n*)&sA[abase];
        ahi.q[1] = *(const uint4n*)&sA[abase + 8];
        alo.q[0] = *(const uint4n*)&sA[abase + 16];
        alo.q[1] = *(const uint4n*)&sA[abase + 24];

        #pragma unroll
        for (int nt = 0; nt < NT; ++nt) {
            const int bbase = (nt * 32 + lane) * 8;
            FragBF bh, bl;
            bh.q[0] = *(const uint4n*)&sBh[bbase];
            bh.q[1] = *(const uint4n*)&sBh[bbase + 4];
            bl.q[0] = *(const uint4n*)&sBl[bbase];
            bl.q[1] = *(const uint4n*)&sBl[bbase + 4];
            acc[nt] = __builtin_amdgcn_wmma_f32_16x16x32_bf16(
                false, ahi.v, false, bh.v, (short)0, acc[nt], false, false);
            acc[nt] = __builtin_amdgcn_wmma_f32_16x16x32_bf16(
                false, ahi.v, false, bl.v, (short)0, acc[nt], false, false);
            acc[nt] = __builtin_amdgcn_wmma_f32_16x16x32_bf16(
                false, alo.v, false, bh.v, (short)0, acc[nt], false, false);
        }
        __syncthreads();
    }

    // --- epilogue: e = tanh(acc + pre), partial score = sum_n e * v[n] ---
    // C/D layout: VGPR v -> row (v + half*8), col n = nt*16 + (lane&15).
    float sp[8];
    #pragma unroll
    for (int v = 0; v < 8; ++v) sp[v] = 0.0f;
    #pragma unroll
    for (int nt = 0; nt < NT; ++nt) {
        int n = nt * 16 + m;
        float vn = (n < H) ? vvec[n] : 0.0f;
        #pragma unroll
        for (int v = 0; v < 8; ++v) {
            int b = wave * 16 + v + half * 8;
            float pn = (n < H) ? pre[b * H + n] : 0.0f;
            sp[v] += fast_tanh(acc[nt][v] + pn) * vn;
        }
    }
    #pragma unroll
    for (int v = 0; v < 8; ++v) {
        sp[v] += __shfl_xor(sp[v], 1, 32);
        sp[v] += __shfl_xor(sp[v], 2, 32);
        sp[v] += __shfl_xor(sp[v], 4, 32);
        sp[v] += __shfl_xor(sp[v], 8, 32);
    }
    if (m == 0) {   // lanes 0 and 16 hold the row totals
        #pragma unroll
        for (int v = 0; v < 8; ++v) {
            int b = wave * 16 + v + half * 8;
            score[b * L + l] = sp[v];
        }
    }
}

// ---------------------------------------------------------------------------
// 4) Masked softmax over L per batch (one block per b).
// ---------------------------------------------------------------------------
__global__ __launch_bounds__(256) void softmax_kernel(
    const float* __restrict__ score, const int* __restrict__ src_len,
    float* __restrict__ attn)
{
    int b = blockIdx.x, tid = threadIdx.x;
    __shared__ float red[256];
    int sl = src_len[b];
    float vals[8];
    float mx = -3.4e38f;
    #pragma unroll
    for (int k = 0; k < 8; ++k) {
        int l = tid + 256 * k;
        float s = (l < sl) ? score[b * L + l] : -1e9f;
        vals[k] = s;
        mx = fmaxf(mx, s);
    }
    red[tid] = mx; __syncthreads();
    for (int s = 128; s > 0; s >>= 1) {
        if (tid < s) red[tid] = fmaxf(red[tid], red[tid + s]);
        __syncthreads();
    }
    mx = red[0]; __syncthreads();
    float sum = 0.0f;
    #pragma unroll
    for (int k = 0; k < 8; ++k) { vals[k] = __expf(vals[k] - mx); sum += vals[k]; }
    red[tid] = sum; __syncthreads();
    for (int s = 128; s > 0; s >>= 1) {
        if (tid < s) red[tid] += red[tid + s];
        __syncthreads();
    }
    float inv = 1.0f / red[0];
    #pragma unroll
    for (int k = 0; k < 8; ++k)
        attn[b * L + tid + 256 * k] = vals[k] * inv;
}

// ---------------------------------------------------------------------------
// 5) context partial sums: grid (b, s); each block covers 256 l's.
//    Deterministic two-stage reduction (no float atomics).
// ---------------------------------------------------------------------------
__global__ __launch_bounds__(320) void ctx_part_kernel(
    const float* __restrict__ enc, const float* __restrict__ attn,
    float* __restrict__ part)
{
    int b = blockIdx.x, s = blockIdx.y, tid = threadIdx.x;
    __shared__ float wsh[256];
    if (tid < 256) wsh[tid] = attn[b * L + s * 256 + tid];
    __syncthreads();
    if (tid < H) {
        float a = 0.0f;
        const float* e = enc + (long)(s * 256) * (B * H) + b * H + tid;
        #pragma unroll 4
        for (int j = 0; j < 256; ++j)
            a = fmaf(wsh[j], e[(long)j * (B * H)], a);
        part[(s * B + b) * H + tid] = a;
    }
}

__global__ void ctx_reduce_kernel(const float* __restrict__ part,
                                  float* __restrict__ ctx)
{
    int i = blockIdx.x * 256 + threadIdx.x;
    if (i < B * H) {
        float a = 0.0f;
        #pragma unroll
        for (int s = 0; s < 8; ++s) a += part[s * (B * H) + i];
        ctx[i] = a;
    }
}

// ---------------------------------------------------------------------------
// Launcher. Inputs (setup_inputs order):
//  0 last_hidden [B,H] f32   1 encoder_outputs [L,B,H] f32
//  2 attn_W [H,2H] f32       3 attn_b [H] f32    4 v [H] f32
//  5 Wp_W (dead)             6 Wp_b (dead)       7 v_p (dead)
//  8 src_len [B] i32
// Output: context [B*H] f32 then attn_weights [B*L] f32.
// ---------------------------------------------------------------------------
extern "C" void kernel_launch(void* const* d_in, const int* in_sizes, int n_in,
                              void* d_out, int out_size, void* d_ws, size_t ws_size,
                              hipStream_t stream)
{
    (void)in_sizes; (void)n_in; (void)out_size; (void)ws_size;
    const float* lh   = (const float*)d_in[0];
    const float* enc  = (const float*)d_in[1];
    const float* aW   = (const float*)d_in[2];
    const float* ab   = (const float*)d_in[3];
    const float* vv   = (const float*)d_in[4];
    const int*   slen = (const int*)d_in[8];

    float* ctx  = (float*)d_out;            // [B*H]
    float* attn = (float*)d_out + B * H;    // [B*L]

    float*    ws    = (float*)d_ws;
    float*    pre   = ws;                                  // 38400 f32
    unsigned* Bh    = (unsigned*)(ws + B * H);             // 48640 dw
    unsigned* Bl    = (unsigned*)(ws + B * H + KC * BFRAG_DW);
    float*    score = ws + B * H + 2 * KC * BFRAG_DW;      // 262144 f32
    float*    part  = ws;   // reused after softmax (pre/Bh/Bl/score all dead)

    pre_kernel<<<B, 320, 0, stream>>>(lh, aW, ab, pre);
    w2prep_kernel<<<(KC * BFRAG_DW + 255) / 256, 256, 0, stream>>>(aW, Bh, Bl);
    energy_score_kernel<<<L, 256, 0, stream>>>(enc, pre, Bh, Bl, vv, score);
    softmax_kernel<<<B, 256, 0, stream>>>(score, slen, attn);
    ctx_part_kernel<<<dim3(B, 8), 320, 0, stream>>>(enc, attn, part);
    ctx_reduce_kernel<<<(B * H + 255) / 256, 256, 0, stream>>>(part, ctx);
}
